// GraphSAGE_85615878078999
// MI455X (gfx1250) — matrix operand order repaired and checked by reading, not compiled, not run on hardware
//
#include <hip/hip_runtime.h>
#include <math.h>

// ---------------------------------------------------------------------------
// GraphSAGE (2-layer, mean aggregation) for MI455X / gfx1250.
// GEMMs via V_WMMA_F32_16X16X32_BF16 (wave32, 16x16 tile per wave).
// ---------------------------------------------------------------------------

typedef __attribute__((ext_vector_type(16))) __bf16 v16bf;
typedef __attribute__((ext_vector_type(8)))  __bf16 v8bf;
typedef __attribute__((ext_vector_type(8)))  float  v8f;

#define IN_F  128
#define HID   128
#define OUTC  16
#define BN_EPS 1e-5f

// ---- A fragment (16x32 bf16, MxK). ISA 7.12.2:
// lanes 0-15 : row M=lane,   halves 0..7 -> K=0..7,  halves 8..15 -> K=16..23
// lanes 16-31: row M=lane-16,halves 0..7 -> K=8..15, halves 8..15 -> K=24..31
// caller passes p = base + row*ld + k0 + 8*(lane>=16); we load [p,p+8) and [p+16,p+24).
__device__ __forceinline__ v16bf load_afrag(const __bf16* p) {
    v8bf lo = *(const v8bf*)(p);
    v8bf hi = *(const v8bf*)(p + 16);
    return __builtin_shufflevector(lo, hi, 0,1,2,3,4,5,6,7,8,9,10,11,12,13,14,15);
}

// ---------------------------------------------------------------------------
__global__ void fill0_kernel(float* p, size_t n) {
    size_t i  = (size_t)blockIdx.x * blockDim.x + threadIdx.x;
    size_t st = (size_t)gridDim.x * blockDim.x;
    for (; i < n; i += st) p[i] = 0.0f;
}

__global__ void cvt_weights_kernel(const float* wl1, const float* wr1,
                                   const float* wl2, const float* wr2,
                                   __bf16* wl1b, __bf16* wr1b,
                                   __bf16* wl2b, __bf16* wr2b) {
    int i = blockIdx.x * blockDim.x + threadIdx.x;   // 16384 threads
    if (i < HID * IN_F) { wl1b[i] = (__bf16)wl1[i]; wr1b[i] = (__bf16)wr1[i]; }
    if (i < OUTC * HID) { wl2b[i] = (__bf16)wl2[i]; wr2b[i] = (__bf16)wr2[i]; }
}

__global__ void degree_kernel(const int* dst, float* cnt, int E) {
    int e = blockIdx.x * blockDim.x + threadIdx.x;
    if (e < E) atomicAdd(&cnt[dst[e]], 1.0f);
}

// one thread per (edge, 4-float chunk): E*32 threads
__global__ void scatter_kernel(const float* feat, const int* src, const int* dst,
                               float* summ, int E) {
    int t = blockIdx.x * blockDim.x + threadIdx.x;
    int e = t >> 5;
    if (e >= E) return;
    int c = (t & 31) * 4;
    int s = src[e], d = dst[e];
    const float4 v = *(const float4*)(feat + (size_t)s * IN_F + c);
    float* o = summ + (size_t)d * IN_F + c;
    atomicAdd(o + 0, v.x); atomicAdd(o + 1, v.y);
    atomicAdd(o + 2, v.z); atomicAdd(o + 3, v.w);
}

// mean = summ / clip(cnt,1); write bf16 agg; optionally also bf16 copy of root feats
__global__ void mean_cvt_kernel(const float* summ, const float* cnt,
                                const float* root, __bf16* aggb, __bf16* rootb,
                                size_t n) {
    size_t i = (size_t)blockIdx.x * blockDim.x + threadIdx.x;
    if (i >= n) return;
    float c = cnt[i >> 7];
    c = c < 1.0f ? 1.0f : c;
    aggb[i] = (__bf16)(summ[i] / c);
    if (rootb) rootb[i] = (__bf16)root[i];
}

// ---- GEMM1: h[N x 128] = agg @ Wl1^T + bl1 + x @ Wr1^T ; also BN stat atomics
__global__ __launch_bounds__(256) void gemm1_kernel(
        const __bf16* __restrict__ aggb, const __bf16* __restrict__ xb,
        const __bf16* __restrict__ wl,   const __bf16* __restrict__ wr,
        const float*  __restrict__ bl,
        float* __restrict__ h, float* __restrict__ bnsum, float* __restrict__ bnsq,
        int N) {
    const int wave = threadIdx.x >> 5;           // 8 waves -> 8 column tiles of H
    const int lane = threadIdx.x & 31;
    const int half = lane >> 4;                  // 0: lanes 0-15, 1: lanes 16-31
    const int lm   = lane & 15;
    const int m0   = blockIdx.x * 16;
    const int n0   = wave * 16;
    if (m0 >= N) return;

    const __bf16* aA = aggb + (size_t)(m0 + lm) * IN_F + half * 8;
    const __bf16* aX = xb   + (size_t)(m0 + lm) * IN_F + half * 8;
    const __bf16* bL = wl   + (size_t)(n0 + lm) * IN_F + half * 8;
    const __bf16* bR = wr   + (size_t)(n0 + lm) * IN_F + half * 8;

    v8f c = {};
#pragma unroll
    for (int k0 = 0; k0 < IN_F; k0 += 32) {
        if (k0 + 32 < IN_F) {                    // global_prefetch_b8
            __builtin_prefetch(aA + k0 + 32, 0, 1);
            __builtin_prefetch(aX + k0 + 32, 0, 1);
        }
        v16bf a1 = load_afrag(aA + k0);
        v16bf b1 = load_afrag(bL + k0);
        c = __builtin_amdgcn_wmma_f32_16x16x32_bf16(false, a1, false, b1,
                                                    (short)0, c, false, false);
        v16bf a2 = load_afrag(aX + k0);
        v16bf b2 = load_afrag(bR + k0);
        c = __builtin_amdgcn_wmma_f32_16x16x32_bf16(false, a2, false, b2,
                                                    (short)0, c, false, false);
    }

    const int n = n0 + lm;                       // output column of this lane
    const float bias = bl[n];
    float s = 0.0f, s2 = 0.0f;
#pragma unroll
    for (int v = 0; v < 8; ++v) {                // rows m0 + v + 8*half
        float val = c[v] + bias;
        h[(size_t)(m0 + v + half * 8) * HID + n] = val;
        s += val; s2 += val * val;
    }
    atomicAdd(&bnsum[n], s);
    atomicAdd(&bnsq[n],  s2);
}

// ---- BN (batch stats) -> PReLU -> residual; also bf16 copy for layer-2 root term
__global__ void bn_prelu_res_kernel(float* h, const float* x,
                                    const float* bnsum, const float* bnsq,
                                    const float* gamma, const float* beta,
                                    const float* aslope, __bf16* hb,
                                    float invN, size_t n) {
    size_t i = (size_t)blockIdx.x * blockDim.x + threadIdx.x;
    if (i >= n) return;
    int j = (int)(i & (HID - 1));
    float mean = bnsum[j] * invN;
    float var  = bnsq[j] * invN - mean * mean;
    float v = h[i];
    v = gamma[j] * (v - mean) * rsqrtf(var + BN_EPS) + beta[j];
    v = v >= 0.0f ? v : aslope[0] * v;           // PReLU shared slope
    v += x[i];                                   // residual (hidden == in)
    h[i]  = v;
    hb[i] = (__bf16)v;
}

// ---- GEMM2 + fused log_softmax: out[N x 16] = agg2 @ Wl2^T + bl2 + h @ Wr2^T
__global__ __launch_bounds__(256) void gemm2_kernel(
        const __bf16* __restrict__ agg2b, const __bf16* __restrict__ hb,
        const __bf16* __restrict__ wl,    const __bf16* __restrict__ wr,
        const float*  __restrict__ bl,
        float* __restrict__ out, int N) {
    const int wave = threadIdx.x >> 5;           // 8 row tiles per block
    const int lane = threadIdx.x & 31;
    const int half = lane >> 4;
    const int lm   = lane & 15;
    const int m0   = (blockIdx.x * 8 + wave) * 16;
    if (m0 >= N) return;

    const __bf16* aA = agg2b + (size_t)(m0 + lm) * HID + half * 8;
    const __bf16* aH = hb    + (size_t)(m0 + lm) * HID + half * 8;
    const __bf16* bL = wl + (size_t)lm * HID + half * 8;   // Wl2: 16 x 128
    const __bf16* bR = wr + (size_t)lm * HID + half * 8;

    v8f c = {};
#pragma unroll
    for (int k0 = 0; k0 < HID; k0 += 32) {
        v16bf a1 = load_afrag(aA + k0);
        v16bf b1 = load_afrag(bL + k0);
        c = __builtin_amdgcn_wmma_f32_16x16x32_bf16(false, a1, false, b1,
                                                    (short)0, c, false, false);
        v16bf a2 = load_afrag(aH + k0);
        v16bf b2 = load_afrag(bR + k0);
        c = __builtin_amdgcn_wmma_f32_16x16x32_bf16(false, a2, false, b2,
                                                    (short)0, c, false, false);
    }

    const float bias = bl[lm];
    // Each register v holds row (m0 + v + 8*half); the 16 class scores of that
    // row live across the 16 lanes of this half-wave -> shfl_xor reduction.
#pragma unroll
    for (int v = 0; v < 8; ++v) {
        float val = c[v] + bias;
        float mx = val;
#pragma unroll
        for (int msk = 8; msk >= 1; msk >>= 1)
            mx = fmaxf(mx, __shfl_xor(mx, msk, 32));
        float se = expf(val - mx);
#pragma unroll
        for (int msk = 8; msk >= 1; msk >>= 1)
            se += __shfl_xor(se, msk, 32);
        out[(size_t)(m0 + v + half * 8) * OUTC + lm] = val - mx - logf(se);
    }
}

// ---------------------------------------------------------------------------
extern "C" void kernel_launch(void* const* d_in, const int* in_sizes, int n_in,
                              void* d_out, int out_size, void* d_ws, size_t ws_size,
                              hipStream_t stream) {
    const float* x     = (const float*)d_in[0];
    const int*   ei    = (const int*)  d_in[1];
    const float* Wl1   = (const float*)d_in[2];
    const float* bl1   = (const float*)d_in[3];
    const float* Wr1   = (const float*)d_in[4];
    const float* Wl2   = (const float*)d_in[5];
    const float* bl2   = (const float*)d_in[6];
    const float* Wr2   = (const float*)d_in[7];
    const float* gamma = (const float*)d_in[8];
    const float* beta  = (const float*)d_in[9];
    const float* a     = (const float*)d_in[10];

    const int N = in_sizes[0] / IN_F;   // 50000
    const int E = in_sizes[1] / 2;      // 800000
    const int* src = ei;
    const int* dst = ei + E;
    const size_t NF = (size_t)N * IN_F;

    // workspace carve-out (256B aligned)
    size_t off = 0;
    auto carve = [&](size_t bytes) -> char* {
        char* p = (char*)d_ws + off;
        off += (bytes + 255) & ~(size_t)255;
        return p;
    };
    float*  summ  = (float*) carve(NF * 4);        // zeroed region start
    float*  cnt   = (float*) carve((size_t)N * 4);
    float*  bnsum = (float*) carve(HID * 4);
    float*  bnsq  = (float*) carve(HID * 4);
    size_t  zeroSpan = off / 4;                    // floats from summ to end of bnsq
    float*  h     = (float*) carve(NF * 4);
    __bf16* aggb  = (__bf16*)carve(NF * 2);
    __bf16* xb    = (__bf16*)carve(NF * 2);
    __bf16* hb    = (__bf16*)carve(NF * 2);
    __bf16* wl1b  = (__bf16*)carve((size_t)HID * IN_F * 2);
    __bf16* wr1b  = (__bf16*)carve((size_t)HID * IN_F * 2);
    __bf16* wl2b  = (__bf16*)carve((size_t)OUTC * HID * 2);
    __bf16* wr2b  = (__bf16*)carve((size_t)OUTC * HID * 2);
    (void)ws_size; (void)n_in; (void)out_size;

    float* out = (float*)d_out;

    const int T = 256;
    // 1. zero summ + cnt + BN stats
    fill0_kernel<<<2048, T, 0, stream>>>(summ, zeroSpan);
    // 2. weights -> bf16
    cvt_weights_kernel<<<(HID * IN_F + T - 1) / T, T, 0, stream>>>(
        Wl1, Wr1, Wl2, Wr2, wl1b, wr1b, wl2b, wr2b);
    // 3. in-degrees
    degree_kernel<<<(E + T - 1) / T, T, 0, stream>>>(dst, cnt, E);
    // 4. scatter-add x into summ
    scatter_kernel<<<((size_t)E * 32 + T - 1) / T, T, 0, stream>>>(x, src, dst, summ, E);
    // 5. mean + bf16 convert (agg1, x)
    mean_cvt_kernel<<<(NF + T - 1) / T, T, 0, stream>>>(summ, cnt, x, aggb, xb, NF);
    // 6. GEMM1 (WMMA) + BN stat accumulation
    gemm1_kernel<<<N / 16, T, 0, stream>>>(aggb, xb, wl1b, wr1b, bl1, h, bnsum, bnsq, N);
    // 7. BN -> PReLU -> residual (+ bf16 copy of h)
    bn_prelu_res_kernel<<<(NF + T - 1) / T, T, 0, stream>>>(
        h, x, bnsum, bnsq, gamma, beta, a, hb, 1.0f / (float)N, NF);
    // 8. re-zero summ for layer 2
    fill0_kernel<<<2048, T, 0, stream>>>(summ, NF);
    // 9. scatter-add h into summ
    scatter_kernel<<<((size_t)E * 32 + T - 1) / T, T, 0, stream>>>(h, src, dst, summ, E);
    // 10. mean + bf16 (agg2); no root copy needed (hb already built)
    mean_cvt_kernel<<<(NF + T - 1) / T, T, 0, stream>>>(summ, cnt, h, aggb, (__bf16*)nullptr, NF);
    // 11. GEMM2 (WMMA) + fused log_softmax
    gemm2_kernel<<<(N / 16 + 7) / 8, T, 0, stream>>>(aggb, hb, wl2b, wr2b, bl2, out, N);
}
